// CrossViewSwapAttention_82051055223341
// MI455X (gfx1250) — matrix operand order repaired and checked by reading, not compiled, not run on hardware
//
#include <hip/hip_runtime.h>
#include <hip/hip_fp16.h>

typedef __attribute__((ext_vector_type(16))) _Float16 v16h;
typedef __attribute__((ext_vector_type(8)))  _Float16 v8h;
typedef __attribute__((ext_vector_type(8)))  float    v8f;
typedef int v4i __attribute__((vector_size(4 * sizeof(int))));

#define AS1 __attribute__((address_space(1)))
#define AS3 __attribute__((address_space(3)))

#if __has_builtin(__builtin_amdgcn_global_load_async_to_lds_b128) && \
    __has_builtin(__builtin_amdgcn_s_wait_asynccnt)
#define HAVE_ASYNC_LDS 1
#else
#define HAVE_ASYNC_LDS 0
#endif

constexpr int NB    = 2;      // batch
constexpr int NCAM  = 6;      // cameras
constexpr int CD    = 128;    // embed dim D (== HD)
constexpr int HQ    = 128, WQ = 128;
constexpr int PIXQ  = HQ * WQ;          // 16384 bev pixels
constexpr int HFD   = 32,  WFD = 32;
constexpr int PIXK  = HFD * WFD;        // 1024 image pixels
constexpr int NHEAD = 4;
constexpr int DHEAD = 32;

__device__ __forceinline__ int laneid() { return threadIdx.x & 31; }

// ---- wave reductions (wave32) ----
__device__ __forceinline__ float rsum16(float v) {
  v += __shfl_xor(v, 1, 32); v += __shfl_xor(v, 2, 32);
  v += __shfl_xor(v, 4, 32); v += __shfl_xor(v, 8, 32);
  return v;
}
__device__ __forceinline__ float rmax16(float v) {
  v = fmaxf(v, __shfl_xor(v, 1, 32)); v = fmaxf(v, __shfl_xor(v, 2, 32));
  v = fmaxf(v, __shfl_xor(v, 4, 32)); v = fmaxf(v, __shfl_xor(v, 8, 32));
  return v;
}
__device__ __forceinline__ float rsum32(float v) {
  v = rsum16(v); v += __shfl_xor(v, 16, 32); return v;
}

// ---- WMMA fragment helpers (layouts per CDNA5 ISA 7.12.2, f16 16x16x32) ----
// A (16xK=32, row-major source, ld in elements):
//   lane<16: row=lane, halves K[0..7] & K[16..23]; lane>=16: row=lane-16, K[8..15] & K[24..31]
__device__ __forceinline__ v16h frag_a_load(const _Float16* p, int ld) {
  int l = laneid();
  const _Float16* r = p + (l & 15) * ld + ((l >> 4) << 3);
  v8h lo = *(const v8h*)r;
  v8h hi = *(const v8h*)(r + 16);
  v16h o;
#pragma unroll
  for (int i = 0; i < 8; ++i) { o[i] = lo[i]; o[i + 8] = hi[i]; }
  return o;
}
// B (K=32 x 16). Source is Bt row-major [N][K] (i.e. column n of B contiguous in K):
//   lane<16: col=lane, K[0..7] & K[16..23]; lane>=16: col=lane-16, K[8..15] & K[24..31]
__device__ __forceinline__ v16h frag_b_load(const _Float16* p, int ld) {
  int l = laneid();
  const _Float16* r = p + (l & 15) * ld + ((l >> 4) << 3);
  v8h lo = *(const v8h*)r;
  v8h hi = *(const v8h*)(r + 16);
  v16h o;
#pragma unroll
  for (int i = 0; i < 8; ++i) { o[i] = lo[i]; o[i + 8] = hi[i]; }
  return o;
}
__device__ __forceinline__ v8f wmma_f16(v16h a, v16h b, v8f c) {
  return __builtin_amdgcn_wmma_f32_16x16x32_f16(false, a, false, b, (short)0, c,
                                                false, false);
}

// ---- weight prep: f32 [K,N] (tr=1) or [N,K] (tr=0)  ->  f16 Wt[N][K] ----
__global__ void k_prep_w(const float* __restrict__ W, _Float16* __restrict__ Wt,
                         int K, int N, int tr) {
  int idx = blockIdx.x * 256 + threadIdx.x;
  if (idx >= K * N) return;
  int k = idx % K, n = idx / K;
  float v = tr ? W[(size_t)k * N + n] : W[(size_t)n * K + k];
  Wt[(size_t)n * K + k] = (_Float16)v;
}

// ---- c_embed[bn][o] = cam_w[o,:] . E_inv[bn][:,3] ----
__global__ void k_setup_cam(const float* __restrict__ E_inv,
                            const float* __restrict__ cam_w,
                            float* __restrict__ c_embed) {
  for (int idx = threadIdx.x; idx < NB * NCAM * CD; idx += 256) {
    int bn = idx >> 7, o = idx & 127;
    float acc = 0.f;
#pragma unroll
    for (int r = 0; r < 4; ++r)
      acc += cam_w[o * 4 + r] * E_inv[bn * 16 + r * 4 + 3];
    c_embed[idx] = acc;
  }
}

// ---- w_embed[o][pix]: BEV world grid embed (V inverse computed analytically) ----
__global__ void k_w_embed(const float* __restrict__ bev_w,
                          const float* __restrict__ bev_b,
                          float* __restrict__ w_embed) {
  int idx = blockIdx.x * 256 + threadIdx.x;       // o*PIXQ + pix
  if (idx >= CD * PIXQ) return;
  int pix = idx & (PIXQ - 1);
  int o = idx >> 14;
  int i = pix >> 7, j = pix & 127;
  float gx = j * (256.f / 127.f);
  float gy = i * (256.f / 127.f);
  float w0 = (128.f - gy) * (1.f / 2.56f);   // world[0]
  float w1 = (128.f - gx) * (1.f / 2.56f);   // world[1]
  w_embed[idx] = bev_w[o * 2] * w0 + bev_w[o * 2 + 1] * w1 + bev_b[o];
}

// ---- normalized camera-ray image embedding, token-major [bn*1024][128] ----
__global__ __launch_bounds__(256) void k_img_embed(
    const float* __restrict__ I_inv, const float* __restrict__ E_inv,
    const float* __restrict__ img_w, const float* __restrict__ c_embed,
    float* __restrict__ img_tok) {
  int l = laneid();
  int p = blockIdx.x * 8 + (threadIdx.x >> 5);    // pixel-token 0..12287
  int pi = p & (PIXK - 1);
  int bn = p >> 10;
  int iy = pi >> 5, ix = pi & 31;
  float px = ix * (512.f / 31.f);
  float py = iy * (512.f / 31.f);
  const float* Ii = I_inv + bn * 9;
  float cam[3];
#pragma unroll
  for (int r = 0; r < 3; ++r)
    cam[r] = Ii[r * 3 + 0] * px + Ii[r * 3 + 1] * py + Ii[r * 3 + 2];
  const float* Ei = E_inv + bn * 16;
  float dv[4];
#pragma unroll
  for (int r = 0; r < 4; ++r)
    dv[r] = Ei[r * 4 + 0] * cam[0] + Ei[r * 4 + 1] * cam[1] +
            Ei[r * 4 + 2] * cam[2] + Ei[r * 4 + 3];
  float e[4]; float ss = 0.f;
#pragma unroll
  for (int k = 0; k < 4; ++k) {
    int c = l + 32 * k;
    float v = img_w[c * 4 + 0] * dv[0] + img_w[c * 4 + 1] * dv[1] +
              img_w[c * 4 + 2] * dv[2] + img_w[c * 4 + 3] * dv[3] -
              c_embed[bn * CD + c];
    e[k] = v; ss += v * v;
  }
  ss = rsum32(ss);
  float inv = 1.f / (sqrtf(ss) + 1e-7f);
  float* po = img_tok + (size_t)p * CD;
#pragma unroll
  for (int k = 0; k < 4; ++k) po[l + 32 * k] = e[k] * inv;
}

// ---- BN(eval) + ReLU on features -> f16 token-major [bn*1024][128] ----
__global__ void k_feat_tok(const float* __restrict__ feat,
                           const float* __restrict__ g, const float* __restrict__ b,
                           const float* __restrict__ m, const float* __restrict__ v,
                           _Float16* __restrict__ out) {
  int idx = blockIdx.x * 256 + threadIdx.x;       // t*128 + c
  if (idx >= NB * NCAM * PIXK * CD) return;
  int c = idx & 127;
  int t = idx >> 7;
  int p = t & (PIXK - 1);
  int bn = t >> 10;
  float f = feat[((size_t)bn * CD + c) * PIXK + p];
  float y = (f - m[c]) * rsqrtf(v[c] + 1e-5f) * g[c] + b[c];
  out[idx] = (_Float16)fmaxf(y, 0.f);
}

// ---- LayerNorm tokens f32[T][128] -> f16[T][128] ----
__global__ __launch_bounds__(256) void k_ln(const float* __restrict__ in,
                                            const float* __restrict__ g,
                                            const float* __restrict__ b,
                                            _Float16* __restrict__ out) {
  int l = laneid();
  size_t t = (size_t)blockIdx.x * 8 + (threadIdx.x >> 5);
  const float* p = in + t * CD;
  float q[4]; float mn = 0.f;
#pragma unroll
  for (int k = 0; k < 4; ++k) { q[k] = p[l + 32 * k]; mn += q[k]; }
  mn = rsum32(mn) * (1.f / 128.f);
  float var = 0.f;
#pragma unroll
  for (int k = 0; k < 4; ++k) { float d = q[k] - mn; var += d * d; }
  var = rsum32(var) * (1.f / 128.f);
  float rs = rsqrtf(var + 1e-5f);
  _Float16* po = out + t * CD;
#pragma unroll
  for (int k = 0; k < 4; ++k) {
    int c = l + 32 * k;
    po[c] = (_Float16)((q[k] - mn) * rs * g[c] + b[c]);
  }
}

// ---- stage-1 query tokens: normalize(w_embed - c_embed) + x, then LN -> f16 ----
__global__ __launch_bounds__(256) void k_make_q1(
    const float* __restrict__ w_embed, const float* __restrict__ c_embed,
    const float* __restrict__ x, const float* __restrict__ g,
    const float* __restrict__ be, _Float16* __restrict__ out) {
  int l = laneid();
  int t = blockIdx.x * 8 + (threadIdx.x >> 5);    // token (b,cam,pix)
  int pix = t & (PIXQ - 1);
  int bn = t >> 14;                                // b*NCAM + cam
  int b = bn / NCAM;
  float e[4], q[4]; float ss = 0.f;
#pragma unroll
  for (int k = 0; k < 4; ++k) {
    int c = l + 32 * k;
    float v = w_embed[(size_t)c * PIXQ + pix] - c_embed[bn * CD + c];
    e[k] = v; ss += v * v;
  }
  ss = rsum32(ss);
  float inv = 1.f / (sqrtf(ss) + 1e-7f);
  float mn = 0.f;
#pragma unroll
  for (int k = 0; k < 4; ++k) {
    int c = l + 32 * k;
    q[k] = e[k] * inv + x[((size_t)b * CD + c) * PIXQ + pix];
    mn += q[k];
  }
  mn = rsum32(mn) * (1.f / 128.f);
  float var = 0.f;
#pragma unroll
  for (int k = 0; k < 4; ++k) { float d = q[k] - mn; var += d * d; }
  var = rsum32(var) * (1.f / 128.f);
  float rs = rsqrtf(var + 1e-5f);
  _Float16* po = out + (size_t)t * CD;
#pragma unroll
  for (int k = 0; k < 4; ++k) {
    int c = l + 32 * k;
    po[c] = (_Float16)((q[k] - mn) * rs * g[c] + be[c]);
  }
}

// ---- generic WMMA GEMM: C[T,N] = act(A[T,K]f16 @ Wt[N,K]^T + bias) (+addsrc) ----
// block 256 = 8 waves (4x2), wave computes 32x32; block tile 128x64
__global__ __launch_bounds__(256) void k_gemm(
    const _Float16* __restrict__ A, const _Float16* __restrict__ Bt,
    const float* __restrict__ bias, const float* __restrict__ addsrc,
    float* __restrict__ Cf, _Float16* __restrict__ Ch,
    int T, int Nn, int K, int act) {
  int wv = threadIdx.x >> 5;
  int m0 = blockIdx.y * 128 + (wv >> 1) * 32;
  int n0 = blockIdx.x * 64 + (wv & 1) * 32;
  // weights are hot across all M-tiles: pull the tile toward the caches early
  __builtin_prefetch(Bt + (size_t)n0 * K, 0, 1);
  __builtin_prefetch(Bt + (size_t)(n0 + 16) * K, 0, 1);
  v8f acc[2][2] = {};
  for (int kk = 0; kk < K; kk += 32) {
    v16h a0 = frag_a_load(A + (size_t)m0 * K + kk, K);
    v16h a1 = frag_a_load(A + (size_t)(m0 + 16) * K + kk, K);
    v16h b0 = frag_b_load(Bt + (size_t)n0 * K + kk, K);
    v16h b1 = frag_b_load(Bt + (size_t)(n0 + 16) * K + kk, K);
    acc[0][0] = wmma_f16(a0, b0, acc[0][0]);
    acc[0][1] = wmma_f16(a0, b1, acc[0][1]);
    acc[1][0] = wmma_f16(a1, b0, acc[1][0]);
    acc[1][1] = wmma_f16(a1, b1, acc[1][1]);
  }
  int l = laneid();
  int cn = l & 15, rb = (l >> 4) << 3;
#pragma unroll
  for (int i = 0; i < 2; ++i)
#pragma unroll
    for (int j = 0; j < 2; ++j) {
      int n = n0 + j * 16 + cn;
      float bv = bias ? bias[n] : 0.f;
#pragma unroll
      for (int r = 0; r < 8; ++r) {
        int m = m0 + i * 16 + rb + r;
        float v = acc[i][j][r] + bv;
        if (act == 1) v = 0.5f * v * (1.f + erff(v * 0.70710678118f)); // exact GELU
        size_t o = (size_t)m * Nn + n;
        if (addsrc) v += addsrc[o];
        if (Cf) Cf[o] = v;
        if (Ch) Ch[o] = (_Float16)v;
      }
    }
}

// ---- fused windowed cross-attention per (patch, head, batch) ----
// K/V (96 keys x 32) staged in LDS via async-to-LDS DMA (ASYNCcnt) when
// available; V transposed in LDS afterwards. sim via single WMMA per tile,
// in-register softmax over 16-lane groups, P re-shaped through LDS for P@V.
__global__ __launch_bounds__(256) void k_attn(
    const _Float16* __restrict__ qh, const _Float16* __restrict__ kh,
    const _Float16* __restrict__ vh, _Float16* __restrict__ oh,
    int qBatchTok, int qCamStrideTok, int tilesPerPatch) {
  __shared__ __attribute__((aligned(16))) _Float16 sK[96][32];
  __shared__ __attribute__((aligned(16))) _Float16 sVt[32][96];
  __shared__ __attribute__((aligned(16))) _Float16 sP[8][16][96];
#if HAVE_ASYNC_LDS
  __shared__ __attribute__((aligned(16))) _Float16 sVr[96][32];
#endif

  int patch = blockIdx.x;            // 0..63  (qx*8+qy)
  int head  = blockIdx.y;            // 0..3
  int b     = blockIdx.z;            // 0..1
  int qx = patch >> 3, qy = patch & 7;
  int tid = threadIdx.x;

  {
    int kidx = (tid >> 1) % 96;      // key 0..95  = cam*16 + ki*4 + kj
    int half = tid & 1;
    int cam = kidx >> 4;
    int rr = kidx & 15;
    int ki = rr >> 2, kj = rr & 3;
    int tk = (b * NCAM + cam) * PIXK + (qx * 4 + ki) * WFD + (qy * 4 + kj);
    const _Float16* kp = kh + (size_t)tk * CD + head * DHEAD + half * 16;
    const _Float16* vp = vh + (size_t)tk * CD + head * DHEAD + half * 16;
#if HAVE_ASYNC_LDS
    if (tid < 192) {
      __builtin_amdgcn_global_load_async_to_lds_b128(
          (AS1 v4i*)kp, (AS3 v4i*)&sK[kidx][half * 16], 0, 0);
      __builtin_amdgcn_global_load_async_to_lds_b128(
          (AS1 v4i*)(kp + 8), (AS3 v4i*)&sK[kidx][half * 16 + 8], 0, 0);
      __builtin_amdgcn_global_load_async_to_lds_b128(
          (AS1 v4i*)vp, (AS3 v4i*)&sVr[kidx][half * 16], 0, 0);
      __builtin_amdgcn_global_load_async_to_lds_b128(
          (AS1 v4i*)(vp + 8), (AS3 v4i*)&sVr[kidx][half * 16 + 8], 0, 0);
    }
    __builtin_amdgcn_s_wait_asynccnt(0);
    __syncthreads();
    if (tid < 192) {                 // LDS->LDS transpose of V
#pragma unroll
      for (int d = 0; d < 16; ++d) sVt[half * 16 + d][kidx] = sVr[kidx][half * 16 + d];
    }
#else
    if (tid < 192) {
      *(v8h*)&sK[kidx][half * 16]     = *(const v8h*)kp;
      *(v8h*)&sK[kidx][half * 16 + 8] = *(const v8h*)(kp + 8);
#pragma unroll
      for (int d = 0; d < 16; ++d) sVt[half * 16 + d][kidx] = vp[d];
    }
#endif
  }
  __syncthreads();

  int wv = tid >> 5;
  int l = laneid();
  int cn = l & 15, rb = (l >> 4) << 3;
  const float scale = 0.17677669529663687f;       // 32^-0.5
  int iters = tilesPerPatch >> 3;
  for (int it = 0; it < iters; ++it) {
    int tile = it * 8 + wv;
    int q0 = tile << 4;
    int cam = q0 >> 8;
    int wi = (q0 & 255) >> 4;
    int pix = (qx * 16 + wi) * WQ + qy * 16;
    int tq0 = b * qBatchTok + cam * qCamStrideTok + pix;

    v16h aq = frag_a_load(qh + (size_t)tq0 * CD + head * DHEAD, CD);
    v8f s[6];
#pragma unroll
    for (int nt = 0; nt < 6; ++nt) {
      v16h bk = frag_b_load(&sK[nt * 16][0], 32);
      v8f z = {};
      s[nt] = wmma_f16(aq, bk, z);
    }
    // softmax over 96 keys (distributed across 16-lane group x 6 tiles)
#pragma unroll
    for (int r = 0; r < 8; ++r) {
      float mx = -1e30f;
#pragma unroll
      for (int nt = 0; nt < 6; ++nt) {
        float v = s[nt][r] * scale; s[nt][r] = v; mx = fmaxf(mx, v);
      }
      mx = rmax16(mx);
      float sum = 0.f;
#pragma unroll
      for (int nt = 0; nt < 6; ++nt) {
        float e = __expf(s[nt][r] - mx); s[nt][r] = e; sum += e;
      }
      sum = rsum16(sum);
      float inv = 1.f / sum;
#pragma unroll
      for (int nt = 0; nt < 6; ++nt)
        sP[wv][rb + r][nt * 16 + cn] = (_Float16)(s[nt][r] * inv);
    }
    __syncthreads();
    // O = P(16x96) @ V(96x32)
    v8f o0 = {}, o1 = {};
#pragma unroll
    for (int kc = 0; kc < 3; ++kc) {
      v16h ap  = frag_a_load(&sP[wv][0][kc * 32], 96);
      v16h bv0 = frag_b_load(&sVt[0][kc * 32], 96);
      v16h bv1 = frag_b_load(&sVt[16][kc * 32], 96);
      o0 = wmma_f16(ap, bv0, o0);
      o1 = wmma_f16(ap, bv1, o1);
    }
    _Float16* op = oh + (size_t)tq0 * CD + head * DHEAD;
#pragma unroll
    for (int r = 0; r < 8; ++r) {
      op[(size_t)(rb + r) * CD + cn]      = (_Float16)o0[r];
      op[(size_t)(rb + r) * CD + 16 + cn] = (_Float16)o1[r];
    }
    __syncthreads();
  }
}

// ---- mean over cameras (+skip) -> a tokens f32 [NB*PIXQ][128] ----
__global__ void k_mean_skip(const _Float16* __restrict__ pout,
                            const float* __restrict__ skip,
                            float* __restrict__ out, int nCams, int skipChanMajor) {
  int idx = blockIdx.x * 256 + threadIdx.x;       // t*128 + o
  if (idx >= NB * PIXQ * CD) return;
  int o = idx & 127;
  int t = idx >> 7;
  int pix = t & (PIXQ - 1);
  int b = t >> 14;
  float acc = 0.f;
  if (nCams == NCAM) {
#pragma unroll
    for (int cam = 0; cam < NCAM; ++cam)
      acc += (float)pout[(((size_t)(b * NCAM + cam)) * PIXQ + pix) * CD + o];
    acc *= (1.f / 6.f);
  } else {
    acc = (float)pout[((size_t)b * PIXQ + pix) * CD + o];
  }
  float s = skipChanMajor ? skip[((size_t)b * CD + o) * PIXQ + pix]
                          : skip[(size_t)t * CD + o];
  out[idx] = acc + s;
}

// ---- final post-LN + transpose to (b, D, H, W) ----
__global__ __launch_bounds__(256) void k_final(const float* __restrict__ in,
                                               const float* __restrict__ g,
                                               const float* __restrict__ b,
                                               float* __restrict__ out) {
  int l = laneid();
  int t = blockIdx.x * 8 + (threadIdx.x >> 5);
  int pix = t & (PIXQ - 1);
  int bb = t >> 14;
  const float* p = in + (size_t)t * CD;
  float q[4]; float mn = 0.f;
#pragma unroll
  for (int k = 0; k < 4; ++k) { q[k] = p[l + 32 * k]; mn += q[k]; }
  mn = rsum32(mn) * (1.f / 128.f);
  float var = 0.f;
#pragma unroll
  for (int k = 0; k < 4; ++k) { float d = q[k] - mn; var += d * d; }
  var = rsum32(var) * (1.f / 128.f);
  float rs = rsqrtf(var + 1e-5f);
#pragma unroll
  for (int k = 0; k < 4; ++k) {
    int c = l + 32 * k;
    out[((size_t)bb * CD + c) * PIXQ + pix] = (q[k] - mn) * rs * g[c] + b[c];
  }
}

extern "C" void kernel_launch(void* const* d_in, const int* in_sizes, int n_in,
                              void* d_out, int out_size, void* d_ws, size_t ws_size,
                              hipStream_t stream) {
  (void)in_sizes; (void)n_in; (void)out_size; (void)ws_size;
  auto in = [&](int i) { return (const float*)d_in[i]; };
  const float* x     = in(0);
  const float* feat  = in(1);
  const float* I_inv = in(2);
  const float* E_inv = in(3);
  const float* img_w = in(4);
  const float* cam_w = in(5);
  const float* bev_w = in(6);
  const float* bev_b = in(7);

  char* base = (char*)d_ws;
  size_t off = 0;
  auto allocB = [&](size_t bytes) -> void* {
    void* p = base + off;
    off = (off + bytes + 255) & ~(size_t)255;
    return p;
  };
  auto aF32 = [&](size_t n) { return (float*)allocB(n * 4); };
  auto aF16 = [&](size_t n) { return (_Float16*)allocB(n * 2); };

  // transposed f16 weights
  _Float16* wt_feat[2] = { aF16(CD * CD), aF16(CD * CD) };
  _Float16* wt_q[2]  = { aF16(CD * CD), aF16(CD * CD) };
  _Float16* wt_k[2]  = { aF16(CD * CD), aF16(CD * CD) };
  _Float16* wt_v[2]  = { aF16(CD * CD), aF16(CD * CD) };
  _Float16* wt_p[2]  = { aF16(CD * CD), aF16(CD * CD) };
  _Float16* wt_m1[2] = { aF16(CD * 2 * CD), aF16(CD * 2 * CD) };
  _Float16* wt_m2[2] = { aF16(CD * 2 * CD), aF16(CD * 2 * CD) };

  const size_t TK  = (size_t)NB * NCAM * PIXK;  // 12288 image tokens
  const size_t TQ1 = (size_t)NB * NCAM * PIXQ;  // 196608 stage-1 q tokens
  const size_t TA  = (size_t)NB * PIXQ;         // 32768 bev tokens
  float*     c_emb   = aF32(NB * NCAM * CD);
  float*     w_emb   = aF32((size_t)CD * PIXQ);
  float*     img_tok = aF32(TK * CD);
  _Float16*  featA   = aF16(TK * CD);
  float*     key_tok = aF32(TK * CD);
  float*     val_tok = aF32(TK * CD);
  _Float16*  khb     = aF16(TK * CD);
  _Float16*  vhb     = aF16(TK * CD);
  _Float16*  LNB     = aF16(TQ1 * CD);   // LN'd tokens (reused)
  _Float16*  QHB     = aF16(TQ1 * CD);   // projected q / proj output (reused)
  _Float16*  OBF     = aF16(TQ1 * CD);   // attention output / mlp hidden (reused)
  float*     AM      = aF32(TA * CD);
  float*     AF1     = aF32(TA * CD);
  float*     AF2     = aF32(TA * CD);

  auto prep = [&](int idx, _Float16* wt, int K, int N, int tr) {
    k_prep_w<<<dim3((K * N + 255) / 256), dim3(256), 0, stream>>>(in(idx), wt, K, N, tr);
  };
  prep(12, wt_feat[0], CD, CD, 0);     // feat_proj.w  [o,c] already N,K
  prep(17, wt_feat[1], CD, CD, 0);     // feat_lin.w
  for (int s = 0; s < 2; ++s) {
    int ab = s ? 32 : 18;
    prep(ab + 2,  wt_q[s], CD, CD, 1);
    prep(ab + 6,  wt_k[s], CD, CD, 1);
    prep(ab + 10, wt_v[s], CD, CD, 1);
    prep(ab + 12, wt_p[s], CD, CD, 1);
    prep(48 + s * 6, wt_m1[s], CD, 2 * CD, 1);      // [128,256] -> Wt[256,128]
    prep(50 + s * 6, wt_m2[s], 2 * CD, CD, 1);      // [256,128] -> Wt[128,256]
  }

  k_setup_cam<<<1, 256, 0, stream>>>(E_inv, cam_w, c_emb);
  k_w_embed<<<(CD * PIXQ) / 256, 256, 0, stream>>>(bev_w, bev_b, w_emb);
  k_img_embed<<<(int)(TK / 8), 256, 0, stream>>>(I_inv, E_inv, img_w, c_emb, img_tok);

  // key_t / val_t (token-major f32)
  k_feat_tok<<<(int)(TK * CD / 256), 256, 0, stream>>>(feat, in(8), in(9), in(10), in(11), featA);
  k_gemm<<<dim3(2, (int)(TK / 128)), 256, 0, stream>>>(featA, wt_feat[0], nullptr,
      img_tok, key_tok, nullptr, (int)TK, CD, CD, 0);
  k_feat_tok<<<(int)(TK * CD / 256), 256, 0, stream>>>(feat, in(13), in(14), in(15), in(16), featA);
  k_gemm<<<dim3(2, (int)(TK / 128)), 256, 0, stream>>>(featA, wt_feat[1], nullptr,
      nullptr, val_tok, nullptr, (int)TK, CD, CD, 0);

  for (int s = 0; s < 2; ++s) {
    int ab = s ? 32 : 18;
    int Tq = s ? (int)TA : (int)TQ1;
    // q tokens -> LN -> f16
    if (s == 0)
      k_make_q1<<<(int)(TQ1 / 8), 256, 0, stream>>>(w_emb, c_emb, x, in(ab + 0), in(ab + 1), LNB);
    else
      k_ln<<<(int)(TA / 8), 256, 0, stream>>>(AF1, in(ab + 0), in(ab + 1), LNB);
    k_gemm<<<dim3(2, Tq / 128), 256, 0, stream>>>(LNB, wt_q[s], in(ab + 3),
        nullptr, nullptr, QHB, Tq, CD, CD, 0);
    // k / v projections
    k_ln<<<(int)(TK / 8), 256, 0, stream>>>(key_tok, in(ab + 4), in(ab + 5), LNB);
    k_gemm<<<dim3(2, (int)(TK / 128)), 256, 0, stream>>>(LNB, wt_k[s], in(ab + 7),
        nullptr, nullptr, khb, (int)TK, CD, CD, 0);
    k_ln<<<(int)(TK / 8), 256, 0, stream>>>(val_tok, in(ab + 8), in(ab + 9), LNB);
    k_gemm<<<dim3(2, (int)(TK / 128)), 256, 0, stream>>>(LNB, wt_v[s], in(ab + 11),
        nullptr, nullptr, vhb, (int)TK, CD, CD, 0);
    // attention (stage 2: q broadcast over cams -> compute once per pixel)
    int qBatch  = s ? PIXQ : NCAM * PIXQ;
    int qCamStr = s ? 0 : PIXQ;
    int tilesPP = s ? 16 : 96;
    k_attn<<<dim3(64, NHEAD, NB), 256, 0, stream>>>(QHB, khb, vhb, OBF,
        qBatch, qCamStr, tilesPP);
    // output projection + mean over cams + skip
    k_gemm<<<dim3(2, Tq / 128), 256, 0, stream>>>(OBF, wt_p[s], in(ab + 13),
        nullptr, nullptr, QHB, Tq, CD, CD, 0);
    k_mean_skip<<<(int)(TA * CD / 256), 256, 0, stream>>>(QHB, s ? AF1 : x, AM,
        s ? 1 : NCAM, s ? 0 : 1);
    // MLP block with residual
    int pg = s ? 52 : 46;
    k_ln<<<(int)(TA / 8), 256, 0, stream>>>(AM, in(pg), in(pg + 1), LNB);
    k_gemm<<<dim3(4, (int)(TA / 128)), 256, 0, stream>>>(LNB, wt_m1[s], in(pg + 3),
        nullptr, nullptr, OBF, (int)TA, 2 * CD, CD, 1);
    float* AFo = s ? AF2 : AF1;
    k_gemm<<<dim3(2, (int)(TA / 128)), 256, 0, stream>>>(OBF, wt_m2[s], in(pg + 5),
        AM, AFo, nullptr, (int)TA, CD, 2 * CD, 0);
  }
  k_final<<<(int)(TA / 8), 256, 0, stream>>>(AF2, in(58), in(59), (float*)d_out);
}
